// DynamicConv2d_36326833389645
// MI455X (gfx1250) — compile-verified, matrix operand order checked by reading, not run on hardware
//
#include <hip/hip_runtime.h>
#include <hip/hip_bf16.h>
#include <math.h>

// ---------------------------------------------------------------------------
// Dims
// ---------------------------------------------------------------------------
#define Bn   16
#define Cn   256
#define Hn   56
#define Wn   56
#define HW   (Hn*Wn)          // 3136
#define Kk   7
#define Gg   4
#define Rr   64               // C/4
#define Mrows 800             // B*49 (pooled rows) + B (mean rows)
#define POOLROWS (Bn*Kk*Kk)   // 784

typedef __attribute__((ext_vector_type(2)))  float     v2f;
typedef __attribute__((ext_vector_type(8)))  float     v8f;
typedef __attribute__((ext_vector_type(16))) _Float16  v16h;

#if defined(__has_builtin)
#if __has_builtin(__builtin_amdgcn_wmma_f32_16x16x4_f32)
#define USE_WMMA_F32 1
#endif
#if __has_builtin(__builtin_amdgcn_global_load_async_to_lds_b32) && \
    __has_builtin(__builtin_amdgcn_s_wait_asynccnt)
#define USE_ASYNC_LDS 1
#endif
#endif
#ifndef USE_WMMA_F32
#define USE_WMMA_F32 0
#endif
#ifndef USE_ASYNC_LDS
#define USE_ASYNC_LDS 0
#endif

// ---------------------------------------------------------------------------
// Stage one (b,c) plane of [56x56] into a zero-halo LDS tile of tw x tw
// (pad = halo). On CDNA5 uses GLOBAL_LOAD_ASYNC_TO_LDS_B32 (ASYNCcnt path,
// no VGPR round-trip); halo zeros go to disjoint LDS addresses, so the only
// ordering needed is s_wait_asynccnt(0) in each wave before the caller's
// __syncthreads().
// ---------------------------------------------------------------------------
#if USE_ASYNC_LDS
typedef __attribute__((address_space(1))) int  gbl_i32;
typedef __attribute__((address_space(3))) int  lds_i32;
#endif

__device__ __forceinline__
void lds_tile_load(float* tile, int tw, int pad, const float* __restrict__ plane)
{
#if USE_ASYNC_LDS
    for (int i = threadIdx.x; i < tw*tw; i += 256) {
        int r = i / tw - pad, q = i % tw - pad;
        if (!(r >= 0 && r < Hn && q >= 0 && q < Wn)) tile[i] = 0.0f;   // halo only
    }
    for (int p = threadIdx.x; p < HW; p += 256) {
        int h = p / Wn, w = p % Wn;
        lds_i32* dst = (lds_i32*)&tile[(h+pad)*tw + (w+pad)];
        gbl_i32* src = (gbl_i32*)&plane[p];
        __builtin_amdgcn_global_load_async_to_lds_b32(src, dst, 0, 0);
    }
    __builtin_amdgcn_s_wait_asynccnt(0);
#else
    for (int i = threadIdx.x; i < tw*tw; i += 256) {
        int r = i / tw - pad, q = i % tw - pad;
        tile[i] = (r >= 0 && r < Hn && q >= 0 && q < Wn) ? plane[r*Wn + q] : 0.0f;
    }
#endif
}

// ---------------------------------------------------------------------------
// Kernel 1: fused dw3x3+BN gate  ->  x_fuse, pooled(7x7), plane mean
// One block per (b,c) plane. 58x58 LDS tile (pad 1), x1 read exactly once.
// ---------------------------------------------------------------------------
__global__ __launch_bounds__(256)
void fuse_pool_kernel(const float* __restrict__ x1,
                      const float* __restrict__ dw1_w, const float* __restrict__ dw1_g,
                      const float* __restrict__ dw1_b, const float* __restrict__ dw1_m,
                      const float* __restrict__ dw1_v,
                      const float* __restrict__ dw2_w, const float* __restrict__ dw2_g,
                      const float* __restrict__ dw2_b, const float* __restrict__ dw2_m,
                      const float* __restrict__ dw2_v,
                      float* __restrict__ xfuse,   // [B*C*HW]
                      float* __restrict__ Xmat)    // [800, 256]
{
    const int bc = blockIdx.x;            // b*C + c
    const int c  = bc % Cn;
    const int b  = bc / Cn;

    __shared__ float tile[58*58];
    __shared__ float cell[49];

    lds_tile_load(tile, 58, 1, x1 + (size_t)bc * HW);
    if (threadIdx.x < 49) cell[threadIdx.x] = 0.0f;
    __syncthreads();

    float w1[9], w2[9];
#pragma unroll
    for (int j = 0; j < 9; ++j) { w1[j] = dw1_w[c*9+j]; w2[j] = dw2_w[c*9+j]; }
    const float s1 = dw1_g[c] * rsqrtf(dw1_v[c] + 1e-5f);
    const float t1 = dw1_b[c] - dw1_m[c] * s1;
    const float s2 = dw2_g[c] * rsqrtf(dw2_v[c] + 1e-5f);
    const float t2 = dw2_b[c] - dw2_m[c] * s2;

    float* outp = xfuse + (size_t)bc * HW;
    for (int p = threadIdx.x; p < HW; p += 256) {
        int h = p / Wn, w = p % Wn;
        float a1 = 0.0f, a2 = 0.0f;
#pragma unroll
        for (int kh = 0; kh < 3; ++kh)
#pragma unroll
            for (int kw = 0; kw < 3; ++kw) {
                float x = tile[(h+kh)*58 + (w+kw)];
                a1 = fmaf(x, w1[kh*3+kw], a1);
                a2 = fmaf(x, w2[kh*3+kw], a2);
            }
        float a = fminf(fmaxf(fmaf(a1, s1, t1), 0.0f), 6.0f);   // relu6(bn1)
        float d = fmaf(a2, s2, t2);                             // bn2
        float f = a * d;
        outp[p] = f;
        atomicAdd(&cell[(h >> 3) * Kk + (w >> 3)], f);          // 8x8 pool cells
    }
    __syncthreads();

    if (threadIdx.x < 49)
        Xmat[(b*49 + threadIdx.x)*Cn + c] = cell[threadIdx.x] * (1.0f/64.0f);
    if (threadIdx.x == 0) {
        float s = 0.0f;
        for (int i = 0; i < 49; ++i) s += cell[i];
        Xmat[(POOLROWS + b)*Cn + c] = s * (1.0f/(float)HW);     // plane mean row
    }
}

// ---------------------------------------------------------------------------
// WMMA GEMMs.  One wave (32 lanes) computes one 16x16 output tile.
// out[m,n] = sum_k A[m,k] * Wt[n,k]  (+ bias, opt. GELU)
// f32 path: V_WMMA_F32_16X16X4_F32 (A 16x4 / B 4x16 per step).
// Fallback: confirmed V_WMMA_F32_16X16X32_F16 with on-the-fly f16 convert.
// ---------------------------------------------------------------------------
__device__ __forceinline__ int f16_kmap(int j, int half) {
    return (j < 8) ? (8*half + j) : (16 + 8*half + (j - 8));
}

template<int KDIM, int NDIM, bool GELU>
__device__ __forceinline__
void wmma_gemm_tile(const float* __restrict__ A,    // [M, KDIM] row-major
                    const float* __restrict__ Wt,   // [NDIM, KDIM] row-major
                    const float* __restrict__ bias, // [NDIM]
                    float* __restrict__ Out,        // [M, NDIM]
                    int tm, int tn)
{
    const int lane = threadIdx.x & 31;
    const int half = lane >> 4;
    const int mr   = lane & 15;
    const int m0   = tm * 16;
    const int n    = tn * 16 + mr;

    v8f acc = {};
#if USE_WMMA_F32
    for (int k = 0; k < KDIM; k += 4) {
        v2f a, bfr;
        const float* ar = A  + (size_t)(m0 + mr) * KDIM + k + 2*half;
        const float* br = Wt + (size_t)n * KDIM + k + 2*half;
        a.x = ar[0]; a.y = ar[1];
        bfr.x = br[0]; bfr.y = br[1];
        acc = __builtin_amdgcn_wmma_f32_16x16x4_f32(
            false, a, false, bfr, (short)0, acc, false, false);
    }
#else
    for (int k = 0; k < KDIM; k += 32) {
        v16h a, bfr;
        const float* ar = A  + (size_t)(m0 + mr) * KDIM + k;
        const float* br = Wt + (size_t)n * KDIM + k;
#pragma unroll
        for (int j = 0; j < 16; ++j) {
            int kk = f16_kmap(j, half);
            a[j]   = (_Float16)ar[kk];
            bfr[j] = (_Float16)br[kk];
        }
        acc = __builtin_amdgcn_wmma_f32_16x16x32_f16(
            false, a, false, bfr, (short)0, acc, false, false);
    }
#endif
    const float bn = bias[n];
#pragma unroll
    for (int v = 0; v < 8; ++v) {
        int m = m0 + v + 8*half;                 // C/D layout: VGPR v -> M = v + 8*half
        float val = acc[v] + bn;
        if (GELU) val = 0.5f * val * (1.0f + erff(val * 0.70710678118f));
        Out[(size_t)m * NDIM + n] = val;
    }
}

__global__ __launch_bounds__(32)
void proj1_gemm(const float* __restrict__ Xmat,  // [800,256]
                const float* __restrict__ pw1,   // [64,256]
                const float* __restrict__ pb1,   // [64]
                float* __restrict__ Hmat)        // [800,64]
{
    const int NT = Rr / 16;                      // 4
    int tm = blockIdx.x / NT, tn = blockIdx.x % NT;
    wmma_gemm_tile<Cn, Rr, true>(Xmat, pw1, pb1, Hmat, tm, tn);
}

__global__ __launch_bounds__(32)
void proj2_gemm(const float* __restrict__ Hmat,  // [800,64]
                const float* __restrict__ pw2,   // [1024,64]
                const float* __restrict__ pb2,   // [1024]
                float* __restrict__ Smat)        // [800,1024]
{
    const int NT = (Cn*Gg) / 16;                 // 64
    int tm = blockIdx.x / NT, tn = blockIdx.x % NT;
    wmma_gemm_tile<Rr, Cn*Gg, false>(Hmat, pw2, pb2, Smat, tm, tn);
}

// ---------------------------------------------------------------------------
// Kernel 4: softmax over G + mix -> dynamic weights [B,C,49] and bias [B,C]
// ---------------------------------------------------------------------------
__global__ __launch_bounds__(256)
void mix_kernel(const float* __restrict__ Smat,     // [800,1024]
                const float* __restrict__ weight1,  // [G,C,7,7]
                const float* __restrict__ bias1,    // [G,C]
                float* __restrict__ wdyn,           // [B,C,49]
                float* __restrict__ bdyn)           // [B,C]
{
    const int totalW = Bn * Cn * 49;
    int idx = blockIdx.x * blockDim.x + threadIdx.x;
    if (idx < totalW) {
        int cell = idx % 49;
        int c    = (idx / 49) % Cn;
        int b    = idx / (49 * Cn);
        const float* srow = Smat + (size_t)(b*49 + cell) * (Cn*Gg);
        float v[Gg], mx = -1e30f;
#pragma unroll
        for (int g = 0; g < Gg; ++g) { v[g] = srow[g*Cn + c]; mx = fmaxf(mx, v[g]); }
        float e[Gg], s = 0.0f;
#pragma unroll
        for (int g = 0; g < Gg; ++g) { e[g] = __expf(v[g] - mx); s += e[g]; }
        float inv = 1.0f / s, acc = 0.0f;
#pragma unroll
        for (int g = 0; g < Gg; ++g)
            acc = fmaf(e[g] * inv, weight1[((size_t)(g*Cn + c))*49 + cell], acc);
        wdyn[idx] = acc;
    } else if (idx < totalW + Bn*Cn) {
        int j = idx - totalW;
        int c = j % Cn, b = j / Cn;
        const float* srow = Smat + (size_t)(POOLROWS + b) * (Cn*Gg);
        float v[Gg], mx = -1e30f;
#pragma unroll
        for (int g = 0; g < Gg; ++g) { v[g] = srow[g*Cn + c]; mx = fmaxf(mx, v[g]); }
        float e[Gg], s = 0.0f;
#pragma unroll
        for (int g = 0; g < Gg; ++g) { e[g] = __expf(v[g] - mx); s += e[g]; }
        float inv = 1.0f / s, acc = 0.0f;
#pragma unroll
        for (int g = 0; g < Gg; ++g)
            acc = fmaf(e[g] * inv, bias1[g*Cn + c], acc);
        bdyn[j] = acc;
    }
}

// ---------------------------------------------------------------------------
// Kernel 5: per-sample dynamic depthwise 7x7 (pad 3) + dynamic bias
// One block per (b,c) plane, 62x62 zero-padded LDS tile (async-staged).
// ---------------------------------------------------------------------------
__global__ __launch_bounds__(256)
void dynconv_kernel(const float* __restrict__ xfuse,
                    const float* __restrict__ wdyn,
                    const float* __restrict__ bdyn,
                    float* __restrict__ out)
{
    const int bc = blockIdx.x;
    __shared__ float tile[62*62];
    __shared__ float wk[49];

    lds_tile_load(tile, 62, 3, xfuse + (size_t)bc * HW);
    if (threadIdx.x < 49) wk[threadIdx.x] = wdyn[(size_t)bc*49 + threadIdx.x];
    __syncthreads();

    const float bias = bdyn[bc];
    float* op = out + (size_t)bc * HW;
    for (int p = threadIdx.x; p < HW; p += 256) {
        int h = p / Wn, w = p % Wn;
        float acc = bias;
#pragma unroll
        for (int kh = 0; kh < Kk; ++kh)
#pragma unroll
            for (int kw = 0; kw < Kk; ++kw)
                acc = fmaf(tile[(h+kh)*62 + (w+kw)], wk[kh*Kk+kw], acc);
        op[p] = acc;
    }
}

// ---------------------------------------------------------------------------
// Launch
// ---------------------------------------------------------------------------
extern "C" void kernel_launch(void* const* d_in, const int* in_sizes, int n_in,
                              void* d_out, int out_size, void* d_ws, size_t ws_size,
                              hipStream_t stream) {
    const float* x1     = (const float*)d_in[0];
    const float* dw1_w  = (const float*)d_in[1];
    const float* dw1_g  = (const float*)d_in[2];
    const float* dw1_b  = (const float*)d_in[3];
    const float* dw1_m  = (const float*)d_in[4];
    const float* dw1_v  = (const float*)d_in[5];
    const float* dw2_w  = (const float*)d_in[6];
    const float* dw2_g  = (const float*)d_in[7];
    const float* dw2_b  = (const float*)d_in[8];
    const float* dw2_m  = (const float*)d_in[9];
    const float* dw2_v  = (const float*)d_in[10];
    const float* weight1= (const float*)d_in[11];
    const float* bias1  = (const float*)d_in[12];
    const float* pw1    = (const float*)d_in[13];
    const float* pb1    = (const float*)d_in[14];
    const float* pw2    = (const float*)d_in[15];
    const float* pb2    = (const float*)d_in[16];
    float* out = (float*)d_out;

    char* ws = (char*)d_ws;
    const size_t XF_SZ   = (size_t)Bn*Cn*HW*sizeof(float);        // 51,380,224
    const size_t XMAT_SZ = (size_t)Mrows*Cn*sizeof(float);        //    819,200
    const size_t HMAT_SZ = (size_t)Mrows*Rr*sizeof(float);        //    204,800
    const size_t SMAT_SZ = (size_t)Mrows*Cn*Gg*sizeof(float);     //  3,276,800
    const size_t WDYN_SZ = (size_t)Bn*Cn*49*sizeof(float);        //    802,816
    float* xfuse = (float*)(ws);
    float* Xmat  = (float*)(ws + XF_SZ);
    float* Hmat  = (float*)(ws + XF_SZ + XMAT_SZ);
    float* Smat  = (float*)(ws + XF_SZ + XMAT_SZ + HMAT_SZ);
    float* wdyn  = (float*)(ws + XF_SZ + XMAT_SZ + HMAT_SZ + SMAT_SZ);
    float* bdyn  = (float*)(ws + XF_SZ + XMAT_SZ + HMAT_SZ + SMAT_SZ + WDYN_SZ);

    // 1) fused gated dwconv + pooling (x1 read exactly once)
    fuse_pool_kernel<<<Bn*Cn, 256, 0, stream>>>(x1,
        dw1_w, dw1_g, dw1_b, dw1_m, dw1_v,
        dw2_w, dw2_g, dw2_b, dw2_m, dw2_v,
        xfuse, Xmat);

    // 2) WMMA GEMM1 + GELU : [800,256] x [256,64]
    proj1_gemm<<<(Mrows/16) * (Rr/16), 32, 0, stream>>>(Xmat, pw1, pb1, Hmat);

    // 3) WMMA GEMM2 : [800,64] x [64,1024]
    proj2_gemm<<<(Mrows/16) * ((Cn*Gg)/16), 32, 0, stream>>>(Hmat, pw2, pb2, Smat);

    // 4) softmax over G + mix -> dynamic weights / bias
    {
        int total = Bn*Cn*49 + Bn*Cn;
        mix_kernel<<<(total + 255)/256, 256, 0, stream>>>(Smat, weight1, bias1, wdyn, bdyn);
    }

    // 5) per-sample dynamic depthwise 7x7 + bias
    dynconv_kernel<<<Bn*Cn, 256, 0, stream>>>(xfuse, wdyn, bdyn, out);
}
